// MHPair_to_MSA_65377992179735
// MI455X (gfx1250) — matrix operand order repaired and checked by reading, not compiled, not run on hardware
//
#include <hip/hip_runtime.h>
#include <hip/hip_bf16.h>

// ---------------- problem constants ----------------
#define S_   128
#define L_   384
#define E_   512
#define C2_  256
#define H_   4
#define DH_  128
#define R_   (S_ * L_)          // 49152 rows (s,l)
#define LN_EPS 1e-5f

typedef __attribute__((ext_vector_type(16))) __bf16 v16bf;
typedef __attribute__((ext_vector_type(8)))  __bf16 v8bf;
typedef __attribute__((ext_vector_type(8)))  float  v8f;

// ---------------- TDM availability probe ----------------
#if defined(__has_builtin)
# if __has_builtin(__builtin_amdgcn_tensor_load_to_lds) && \
     __has_builtin(__builtin_amdgcn_s_wait_tensorcnt)
#  define USE_TDM 1
# endif
#endif
#ifndef USE_TDM
# define USE_TDM 0
#endif
#if USE_TDM && defined(__clang_major__) && (__clang_major__ >= 23)
# define TDM_ARGS6 1
#else
# define TDM_ARGS6 0
#endif

#if USE_TDM
typedef unsigned int       u32;
typedef unsigned long long u64;
typedef __attribute__((ext_vector_type(4))) u32 u32x4;
typedef __attribute__((ext_vector_type(8))) int i32x8;
typedef __attribute__((ext_vector_type(4))) int i32x4;

// Issue one TDM 2D tile load (bf16 elements): dim0 = contiguous K run,
// dim1 = rows with stride `stride0_elems`. Tensor dims == tile dims
// (tiles never go OOB in this problem), D# per CDNA5 ISA ch.8.
__device__ __forceinline__
void tdm_load_2d(u32 lds_addr, const void* gptr,
                 u32 dim0_elems, u32 dim1_rows, u64 stride0_elems) {
  const u64 ga = (u64)(uintptr_t)gptr;
  u32x4 g0;
  g0[0] = 1u;                                              // count=1 (valid)
  g0[1] = lds_addr;                                        // LDS byte address
  g0[2] = (u32)ga;                                         // global_addr lo
  g0[3] = (u32)((ga >> 32) & 0x01FFFFFFu) | (2u << 30);    // addr hi | type=2
  i32x8 g1;
  g1[0] = (int)(1u << 16);                                 // data_size=1 (2B)
  g1[1] = (int)((dim0_elems & 0xFFFFu) << 16);             // tensor_dim0 lo16
  g1[2] = (int)(((dim0_elems >> 16) & 0xFFFFu) |
                ((dim1_rows & 0xFFFFu) << 16));            // td0 hi | td1 lo
  g1[3] = (int)(((dim1_rows >> 16) & 0xFFFFu) |
                ((dim0_elems & 0xFFFFu) << 16));           // td1 hi | tile_dim0
  g1[4] = (int)(dim1_rows & 0xFFFFu);                      // tile_dim1 (td2=0)
  g1[5] = (int)(u32)(stride0_elems & 0xFFFFFFFFu);         // dim0 stride lo
  g1[6] = (int)(u32)((stride0_elems >> 32) & 0xFFFFu);     // dim0 stride hi
  g1[7] = 0;
  i32x4 z4 = {};
#if TDM_ARGS6
  i32x8 z8 = {};
  __builtin_amdgcn_tensor_load_to_lds(g0, g1, z4, z4, z8, 0);
#else
  __builtin_amdgcn_tensor_load_to_lds(g0, g1, z4, z4, 0);
#endif
}
#endif  // USE_TDM

// =====================================================================
// 1) LayerNorm over E=512 on x, emit bf16 msa  (one block per row)
// =====================================================================
__global__ __launch_bounds__(256)
void ln_msa_kernel(const float* __restrict__ x,
                   const float* __restrict__ g,
                   const float* __restrict__ b,
                   __bf16* __restrict__ out) {
  const int row = blockIdx.x;
  const int tid = threadIdx.x;
  const float* xr = x + (size_t)row * E_;
  float v0 = xr[tid];
  float v1 = xr[tid + 256];

  __shared__ float red[256];
  red[tid] = v0 + v1;
  __syncthreads();
  #pragma unroll
  for (int off = 128; off > 0; off >>= 1) {
    if (tid < off) red[tid] += red[tid + off];
    __syncthreads();
  }
  const float mu = red[0] * (1.0f / E_);
  __syncthreads();

  const float d0 = v0 - mu, d1 = v1 - mu;
  red[tid] = d0 * d0 + d1 * d1;
  __syncthreads();
  #pragma unroll
  for (int off = 128; off > 0; off >>= 1) {
    if (tid < off) red[tid] += red[tid + off];
    __syncthreads();
  }
  const float rs = rsqrtf(red[0] * (1.0f / E_) + LN_EPS);

  __bf16* orow = out + (size_t)row * E_;
  orow[tid]       = (__bf16)(d0 * rs * g[tid]       + b[tid]);
  orow[tid + 256] = (__bf16)(d1 * rs * g[tid + 256] + b[tid + 256]);
}

// =====================================================================
// 2) Repack weights to bf16 COLUMN-MAJOR B operands:
//    WvC[h*Dh+d][e] = Wv[h][e][d]   (B of GEMM-5, col-major)
//    WoC[o][e]      = Wo[o][e]      (B of GEMM-7, col-major == Wo itself)
// =====================================================================
__global__ __launch_bounds__(256)
void wprep_kernel(const float* __restrict__ Wv,
                  const float* __restrict__ Wo,
                  __bf16* __restrict__ WvC,
                  __bf16* __restrict__ WoC) {
  const int idx = blockIdx.x * 256 + threadIdx.x;   // 0 .. 512*512-1
  const int col = idx / E_;          // h*Dh + d
  const int e   = idx % E_;
  const int h   = col / DH_;
  const int d   = col % DH_;
  WvC[idx] = (__bf16)Wv[((size_t)h * E_ + e) * DH_ + d];
  WoC[idx] = (__bf16)Wo[idx];
}

// =====================================================================
// 3) pair symmetrize + LayerNorm(C=256) + per-head dot with Wp
// =====================================================================
__global__ __launch_bounds__(256)
void pair_scores_kernel(const float* __restrict__ pair,
                        const float* __restrict__ g,
                        const float* __restrict__ b,
                        const float* __restrict__ Wp,
                        float* __restrict__ scores,
                        float scaling) {
  const int ij = blockIdx.x;
  const int i = ij / L_, j = ij % L_;
  const int c = threadIdx.x;

  const float v = 0.5f * (pair[((size_t)i * L_ + j) * C2_ + c] +
                          pair[((size_t)j * L_ + i) * C2_ + c]);

  __shared__ float red[256];
  red[c] = v;
  __syncthreads();
  #pragma unroll
  for (int off = 128; off > 0; off >>= 1) {
    if (c < off) red[c] += red[c + off];
    __syncthreads();
  }
  const float mu = red[0] * (1.0f / C2_);
  __syncthreads();

  const float d = v - mu;
  red[c] = d * d;
  __syncthreads();
  #pragma unroll
  for (int off = 128; off > 0; off >>= 1) {
    if (c < off) red[c] += red[c + off];
    __syncthreads();
  }
  const float rs = rsqrtf(red[0] * (1.0f / C2_) + LN_EPS);
  __syncthreads();

  const float t = d * rs * g[c] + b[c];

  __shared__ float4 red4[256];
  float4 p;
  p.x = t * Wp[c];
  p.y = t * Wp[C2_ + c];
  p.z = t * Wp[2 * C2_ + c];
  p.w = t * Wp[3 * C2_ + c];
  red4[c] = p;
  __syncthreads();
  #pragma unroll
  for (int off = 128; off > 0; off >>= 1) {
    if (c < off) {
      float4 a = red4[c], q = red4[c + off];
      a.x += q.x; a.y += q.y; a.z += q.z; a.w += q.w;
      red4[c] = a;
    }
    __syncthreads();
  }
  if (c == 0) {
    const size_t base = (size_t)i * L_ + j;
    const size_t HL = (size_t)L_ * L_;
    float4 r = red4[0];
    scores[base]          = scaling * r.x;
    scores[HL + base]     = scaling * r.y;
    scores[2 * HL + base] = scaling * r.z;
    scores[3 * HL + base] = scaling * r.w;
  }
}

// =====================================================================
// 4) masked softmax over j (row length 384), emit bf16 attn
// =====================================================================
__global__ __launch_bounds__(128)
void softmax_kernel(const float* __restrict__ scores,
                    const int* __restrict__ mask,
                    __bf16* __restrict__ attn) {
  const int hi = blockIdx.x;
  const int h = hi / L_, i = hi % L_;
  const int tid = threadIdx.x;
  const float* srow = scores + ((size_t)h * L_ + i) * L_;
  const int mi = mask[i];

  float vals[3];
  #pragma unroll
  for (int k = 0; k < 3; ++k) {
    const int j = tid + k * 128;
    vals[k] = (mi * mask[j]) ? srow[j] : -3.0e38f;
  }

  __shared__ float red[128];
  float mx = fmaxf(vals[0], fmaxf(vals[1], vals[2]));
  red[tid] = mx;
  __syncthreads();
  #pragma unroll
  for (int off = 64; off > 0; off >>= 1) {
    if (tid < off) red[tid] = fmaxf(red[tid], red[tid + off]);
    __syncthreads();
  }
  mx = red[0];
  __syncthreads();

  float e[3], s = 0.0f;
  #pragma unroll
  for (int k = 0; k < 3; ++k) { e[k] = __expf(vals[k] - mx); s += e[k]; }
  red[tid] = s;
  __syncthreads();
  #pragma unroll
  for (int off = 64; off > 0; off >>= 1) {
    if (tid < off) red[tid] += red[tid + off];
    __syncthreads();
  }
  const float inv = 1.0f / red[0];

  __bf16* arow = attn + ((size_t)h * L_ + i) * L_;
  #pragma unroll
  for (int k = 0; k < 3; ++k) arow[tid + k * 128] = (__bf16)(e[k] * inv);
}

// =====================================================================
// 5) batched bf16 WMMA GEMM:  C = A(MxK) * Bcol(NxK, col-major) [+resid]
//    Block tile 128x128, BK=64, 128 threads = 4 waves (2x2); each wave
//    owns a 64x64 region = 4x4 accumulators -> 1 ds_load_b128 per WMMA
//    (4x operand reuse), 32 WMMA / wave / K-iteration.
//    Double-buffered 64 KB LDS tiles filled by the Tensor Data Mover
//    (tensor_load_to_lds + s_wait_tensorcnt) when available, else b128.
//    OUTMODE: 0 = f32 row-major, 1 = bf16 row-major,
//             2 = bf16 transposed V-store  VT[s][col][j], row=(s,j),
//                 packed 8-consecutive-j b128 stores (s uniform: 128|L_).
// =====================================================================
template <int OUTMODE, bool RESID>
__global__ __launch_bounds__(128)
void gemm_bf16_kernel(const __bf16* __restrict__ A,
                      const __bf16* __restrict__ B,
                      void* __restrict__ Cout,
                      const float* __restrict__ resid,
                      int K, int lda, int ldb, int ldc,
                      int sdiv,
                      long sA_h, long sB_s, long sB_h, long sC_s, long sC_h) {
  const int tid  = threadIdx.x;
  const int lane = tid & 31;
  const int wave = tid >> 5;            // 0..3
  const int wm   = wave >> 1;           // 0..1  (M dir, 64 rows each)
  const int wn   = wave & 1;            // 0..1  (N dir, 64 cols each)
  const int mblk = blockIdx.y * 128;
  const int nblk = blockIdx.x * 128;

  const int bz = blockIdx.z;
  const int hb = bz / sdiv;
  const int sb = bz % sdiv;
  const size_t offA = (size_t)hb * sA_h;
  const size_t offB = (size_t)sb * sB_s + (size_t)hb * sB_h;
  const size_t offC = (size_t)sb * sC_s + (size_t)hb * sC_h;

  __shared__ __align__(128) __bf16 sA[2][128 * 64];   // [m][k] row-major
  __shared__ __align__(128) __bf16 sB[2][128 * 64];   // [n][k] col-major

  v8f acc[4][4] = {};

#if USE_TDM
  // one wave drives the DMA engine; EXEC ignored by TENSOR ops
  #define STAGE_TILES(bufi, kk)                                               \
    do {                                                                      \
      if (tid < 32) {                                                         \
        tdm_load_2d((u32)(uintptr_t)&sA[(bufi)][0],                           \
                    A + offA + (size_t)mblk * lda + (kk), 64u, 128u,          \
                    (u64)(unsigned)lda);                                      \
        tdm_load_2d((u32)(uintptr_t)&sB[(bufi)][0],                           \
                    B + offB + (size_t)nblk * ldb + (kk), 64u, 128u,          \
                    (u64)(unsigned)ldb);                                      \
      }                                                                       \
    } while (0)
  #define STAGE_WAIT()                                                        \
    do { if (tid < 32) __builtin_amdgcn_s_wait_tensorcnt(0); } while (0)
#else
  #define STAGE_TILES(bufi, kk)                                               \
    do {                                                                      \
      _Pragma("unroll")                                                       \
      for (int it = 0; it < 8; ++it) {       /* A: 8192 elems, 1024 chunks */ \
        const int cid = tid + it * 128;                                       \
        const int r_  = cid >> 3;                                             \
        const int kc_ = (cid & 7) << 3;                                       \
        const __bf16* gA_ = A + offA +                                        \
            (size_t)(mblk + r_) * lda + kc_ + (kk);                           \
        *(uint4*)(&sA[(bufi)][r_ * 64 + kc_]) = *(const uint4*)gA_;           \
      }                                                                       \
      _Pragma("unroll")                                                       \
      for (int it = 0; it < 8; ++it) {       /* B: 8192 elems, 1024 chunks */ \
        const int cid = tid + it * 128;                                       \
        const int n_  = cid >> 3;                                             \
        const int kc_ = (cid & 7) << 3;                                       \
        const __bf16* gB_ = B + offB + (size_t)(nblk + n_) * ldb + kc_ + (kk);\
        *(uint4*)(&sB[(bufi)][n_ * 64 + kc_]) = *(const uint4*)gB_;           \
      }                                                                       \
    } while (0)
  #define STAGE_WAIT() ((void)0)
#endif

  STAGE_TILES(0, 0);
  STAGE_WAIT();
  __syncthreads();

  int buf = 0;
  for (int k0 = 0; k0 < K; k0 += 64) {
    const bool nxt = (k0 + 64) < K;
    if (nxt) STAGE_TILES(buf ^ 1, k0 + 64);   // DMA next tile during compute

    const int rsel = lane & 15;
    const int akh  = (lane < 16) ? 0 : 16;    // A: K halves 0..15 / 16..31
    const int bkb  = (lane < 16) ? 0 : 16;    // B: K halves 0..15 / 16..31

    #pragma unroll
    for (int ks = 0; ks < 2; ++ks) {          // two K substeps of 32
      const int kof = ks * 32;
      v16bf aF[4], bF[4];
      #pragma unroll
      for (int tm = 0; tm < 4; ++tm) {
        const int r = wm * 64 + tm * 16 + rsel;
        v8bf lo = *(const v8bf*)(&sA[buf][r * 64 + kof + akh]);
        v8bf hi = *(const v8bf*)(&sA[buf][r * 64 + kof + akh + 8]);
        aF[tm] = __builtin_shufflevector(lo, hi,
                 0, 1, 2, 3, 4, 5, 6, 7, 8, 9, 10, 11, 12, 13, 14, 15);
      }
      #pragma unroll
      for (int tn = 0; tn < 4; ++tn) {
        const int n = wn * 64 + tn * 16 + rsel;
        bF[tn] = *(const v16bf*)(&sB[buf][n * 64 + kof + bkb]);
      }
      #pragma unroll
      for (int tm = 0; tm < 4; ++tm)
        #pragma unroll
        for (int tn = 0; tn < 4; ++tn)
          acc[tm][tn] = __builtin_amdgcn_wmma_f32_16x16x32_bf16(
              false, aF[tm], false, bF[tn],
              (short)0, acc[tm][tn], false, false);
    }

    if (nxt) STAGE_WAIT();
    __syncthreads();
    buf ^= 1;
  }
  #undef STAGE_TILES
  #undef STAGE_WAIT

  // ---- store C (f32 16x16 layout: VGPR r -> M = r + 8*(lane>=16), N = lane%16)
  const int rowh = (lane < 16) ? 0 : 8;
  const int coll = lane & 15;

  if constexpr (OUTMODE == 2) {
    // block-uniform s / jbase (128 | L_ so the whole block shares one s)
    const int s     = mblk / L_;
    const int jbase = mblk - s * L_;
    __bf16* C = (__bf16*)Cout;
    #pragma unroll
    for (int tm = 0; tm < 4; ++tm) {
      #pragma unroll
      for (int tn = 0; tn < 4; ++tn) {
        const int j0  = jbase + wm * 64 + tm * 16 + rowh;   // 8 consecutive j
        const int nb2 = nblk + wn * 64 + tn * 16 + coll;
        __bf16 tmp[8];
        #pragma unroll
        for (int r = 0; r < 8; ++r) tmp[r] = (__bf16)acc[tm][tn][r];
        const size_t ci = (size_t)s * ((size_t)E_ * L_) + (size_t)nb2 * L_ + j0;
        *(uint4*)(&C[ci]) = *(const uint4*)tmp;             // packed b128 store
      }
    }
  } else {
    #pragma unroll
    for (int tm = 0; tm < 4; ++tm) {
      #pragma unroll
      for (int tn = 0; tn < 4; ++tn) {
        const int mbase = mblk + wm * 64 + tm * 16 + rowh;
        const int nb2   = nblk + wn * 64 + tn * 16 + coll;
        const size_t cibase = offC + (size_t)mbase * ldc + nb2;
        #pragma unroll
        for (int r = 0; r < 8; ++r) {
          const size_t ci = cibase + (size_t)r * ldc;
          float val = acc[tm][tn][r];
          if (RESID) val += resid[ci];
          if constexpr (OUTMODE == 1) ((__bf16*)Cout)[ci] = (__bf16)val;
          else                        ((float*)Cout)[ci]  = val;
        }
      }
    }
  }
}

// =====================================================================
// launcher
// =====================================================================
extern "C" void kernel_launch(void* const* d_in, const int* in_sizes, int n_in,
                              void* d_out, int out_size, void* d_ws, size_t ws_size,
                              hipStream_t stream) {
  const float* x        = (const float*)d_in[0];
  const float* pairf    = (const float*)d_in[1];
  const int*   mask     = (const int*)d_in[2];
  const float* ln_msa_g = (const float*)d_in[3];
  const float* ln_msa_b = (const float*)d_in[4];
  const float* ln_pr_g  = (const float*)d_in[5];
  const float* ln_pr_b  = (const float*)d_in[6];
  const float* Wp       = (const float*)d_in[7];
  const float* Wv       = (const float*)d_in[8];
  const float* Wo       = (const float*)d_in[9];
  float* out = (float*)d_out;

  char* ws = (char*)d_ws;
  size_t o = 0;
  auto alloc = [&](size_t bytes) -> void* {
    o = (o + 255) & ~(size_t)255;
    void* p = ws + o;
    o += bytes;
    return p;
  };

  __bf16* msaB  = (__bf16*)alloc((size_t)R_ * E_ * 2);    // LN(x) bf16
  __bf16* VT    = (__bf16*)alloc((size_t)R_ * E_ * 2);    // V, [s][hd][j]
  __bf16* midB  = (__bf16*)alloc((size_t)R_ * E_ * 2);    // attn@V bf16
  __bf16* WvC   = (__bf16*)alloc((size_t)E_ * E_ * 2);    // col-major
  __bf16* WoC   = (__bf16*)alloc((size_t)E_ * E_ * 2);    // col-major
  float*  scores= (float*) alloc((size_t)H_ * L_ * L_ * 4);
  __bf16* attnB = (__bf16*)alloc((size_t)H_ * L_ * L_ * 2);

  // 1) msa = LN(x) -> bf16
  ln_msa_kernel<<<R_, 256, 0, stream>>>(x, ln_msa_g, ln_msa_b, msaB);

  // 2) weight repack (col-major B operands)
  wprep_kernel<<<(E_ * E_) / 256, 256, 0, stream>>>(Wv, Wo, WvC, WoC);

  // 3) pair LN + per-head scores, scaling = Dh^-0.5 / sqrt(L)
  const float scaling = (1.0f / sqrtf((float)DH_)) / sqrtf((float)L_);
  pair_scores_kernel<<<L_ * L_, 256, 0, stream>>>(pairf, ln_pr_g, ln_pr_b, Wp,
                                                  scores, scaling);

  // 4) masked softmax -> bf16 attn
  softmax_kernel<<<H_ * L_, 128, 0, stream>>>(scores, mask, attnB);

  // 5) V = msa @ WvC^T : [49152,512] x [512,512], stored transposed VT
  gemm_bf16_kernel<2, false><<<dim3(E_ / 128, R_ / 128, 1), 128, 0, stream>>>(
      msaB, WvC, VT, nullptr,
      /*K=*/E_, /*lda=*/E_, /*ldb=*/E_, /*ldc=*/E_,
      /*sdiv=*/1, 0, 0, 0, 0, 0);

  // 6) mid[s,i,h*Dh+d] = sum_j attn[h,i,j] * VT[s][h*Dh+d][j]
  //    batch z = h*S + s : 512 batches of 384x128x384, B col-major
  gemm_bf16_kernel<1, false><<<dim3(1, L_ / 128, H_ * S_), 128, 0, stream>>>(
      attnB, VT, midB, nullptr,
      /*K=*/L_, /*lda=*/L_, /*ldb=*/L_, /*ldc=*/E_,
      /*sdiv=*/S_,
      /*sA_h=*/(long)L_ * L_,
      /*sB_s=*/(long)E_ * L_, /*sB_h=*/(long)DH_ * L_,
      /*sC_s=*/(long)L_ * E_, /*sC_h=*/(long)DH_);

  // 7) y = mid @ WoC^T + x : [49152,512] x [512,512] with f32 residual
  gemm_bf16_kernel<0, true><<<dim3(E_ / 128, R_ / 128, 1), 128, 0, stream>>>(
      midB, WoC, out, x,
      /*K=*/E_, /*lda=*/E_, /*ldb=*/E_, /*ldc=*/E_,
      /*sdiv=*/1, 0, 0, 0, 0, 0);
}
